// CategorySpecificLinear_2723009266339
// MI455X (gfx1250) — matrix operand order repaired and checked by reading, not compiled, not run on hardware
//
#include <hip/hip_runtime.h>

// CategorySpecificLinear for MI455X (gfx1250, wave32, WMMA).
// out[b,t,h] = sum_i x[b,t,i] * W[cat[b],i,h] + bias[cat[b],h]
// f32 emulated via bf16 hi/lo split: x*w ~= xh*wh + xh*wl + xl*wh (3x v_wmma_f32_16x16x32_bf16).
// Double-buffered LDS pipeline: global loads for tile k+1 overlap WMMA on tile k.

typedef __bf16 bf16;
typedef __attribute__((ext_vector_type(16))) __bf16 v16bf;
typedef __attribute__((ext_vector_type(8)))  __bf16 v8bf;
typedef __attribute__((ext_vector_type(4)))  __bf16 v4bf;
typedef __attribute__((ext_vector_type(8)))  float  v8f;

#define DIM_I 1024
#define DIM_H 4096
#define DIM_T 256
#define DIM_B 64

#define BM 128
#define BN 128
#define BK 32
#define NK (DIM_I / BK)
#define LDA 40   // padded bf16 stride (80B) -> conflict-free b128 fragment loads
#define LDB 40

static __device__ __forceinline__ v16bf cat16(v8bf a, v8bf b) {
    return __builtin_shufflevector(a, b, 0,1,2,3,4,5,6,7,8,9,10,11,12,13,14,15);
}

static __device__ __forceinline__ float f4get(const float4& v, int j) {
    switch (j) { case 0: return v.x; case 1: return v.y; case 2: return v.z; default: return v.w; }
}

static __device__ __forceinline__ void split4(const float4 v, v4bf* hi, v4bf* lo) {
    bf16 h0 = (bf16)v.x, h1 = (bf16)v.y, h2 = (bf16)v.z, h3 = (bf16)v.w;
    *hi = (v4bf){h0, h1, h2, h3};
    *lo = (v4bf){(bf16)(v.x - (float)h0), (bf16)(v.y - (float)h1),
                 (bf16)(v.z - (float)h2), (bf16)(v.w - (float)h3)};
}

__global__ __launch_bounds__(256)
void cat_linear_wmma(const float* __restrict__ x,
                     const int*   __restrict__ cat_ids,
                     const float* __restrict__ W,
                     const float* __restrict__ bias,
                     float*       __restrict__ out) {
    // Double-buffered hi/lo bf16 tiles: A = x tile [BM][BK] row-major, B = W tile [BN][BK] transposed.
    __shared__ bf16 As_hi[2][BM * LDA];
    __shared__ bf16 As_lo[2][BM * LDA];
    __shared__ bf16 Bs_hi[2][BN * LDB];
    __shared__ bf16 Bs_lo[2][BN * LDB];

    const int tid   = threadIdx.x;
    const int lane  = tid & 31;
    const int wave  = tid >> 5;
    const int waveM = wave & 3;   // 4 waves along M (32 rows each)
    const int waveN = wave >> 2;  // 2 waves along N (64 cols each)

    const int b   = blockIdx.z;
    const int t0  = blockIdx.y * BM;
    const int h0  = blockIdx.x * BN;
    const int cat = cat_ids[b];

    const float* xA = x + ((size_t)b * DIM_T + t0) * DIM_I;     // [BM][DIM_I]
    const float* Wb = W + (size_t)cat * DIM_I * DIM_H + h0;     // [DIM_I][DIM_H] at col h0

    // Per-thread staging coordinates.
    const int aCol  = (tid & 7) * 4;    // K offset in A tile (4 chunks, rows aRow0 + 32*i)
    const int aRow0 = tid >> 3;
    const int bCol  = (tid & 31) * 4;   // H offset in B tile (one 4x4 f32 block per thread)
    const int bRow0 = (tid >> 5) * 4;   // K row base

    float4 ra[4], rb[4];
    auto load_tile = [&](int kt) {
#pragma unroll
        for (int i = 0; i < 4; ++i)
            ra[i] = *(const float4*)(xA + (size_t)(aRow0 + 32 * i) * DIM_I + kt + aCol);
#pragma unroll
        for (int j = 0; j < 4; ++j)
            rb[j] = *(const float4*)(Wb + (size_t)(kt + bRow0 + j) * DIM_H + bCol);
    };
    auto store_tile = [&](int buf) {
#pragma unroll
        for (int i = 0; i < 4; ++i) {            // A: row-major [m][k], packed v4bf (b64 stores)
            v4bf hi, lo;
            split4(ra[i], &hi, &lo);
            const int base = (aRow0 + 32 * i) * LDA + aCol;
            *(v4bf*)&As_hi[buf][base] = hi;
            *(v4bf*)&As_lo[buf][base] = lo;
        }
#pragma unroll
        for (int j = 0; j < 4; ++j) {            // B: register-transpose 4x4 -> [h][k], b64 stores
            float4 c;
            c.x = f4get(rb[0], j); c.y = f4get(rb[1], j);
            c.z = f4get(rb[2], j); c.w = f4get(rb[3], j);
            v4bf hi, lo;
            split4(c, &hi, &lo);
            const int base = (bCol + j) * LDB + bRow0;
            *(v4bf*)&Bs_hi[buf][base] = hi;
            *(v4bf*)&Bs_lo[buf][base] = lo;
        }
    };

    v8f acc[2][4];
    const v8f zero = {0.f, 0.f, 0.f, 0.f, 0.f, 0.f, 0.f, 0.f};
#pragma unroll
    for (int mi = 0; mi < 2; ++mi)
#pragma unroll
        for (int ni = 0; ni < 4; ++ni) acc[mi][ni] = zero;

    const int halfsel = lane >> 4;  // 0: lanes 0-15, 1: lanes 16-31
    const int l16     = lane & 15;

    auto compute = [&](int buf) {
        // A fragments: 16-bit A 16x32 layout (lane<16: K 0..7 & 16..23; lane>=16: +8)
        v16bf ahi[2], alo[2];
#pragma unroll
        for (int mi = 0; mi < 2; ++mi) {
            const int row = waveM * 32 + mi * 16 + l16;
            const int k0  = halfsel * 8;
            const bf16* pa = &As_hi[buf][row * LDA + k0];
            ahi[mi] = cat16(*(const v8bf*)pa, *(const v8bf*)(pa + 16));
            const bf16* pl = &As_lo[buf][row * LDA + k0];
            alo[mi] = cat16(*(const v8bf*)pl, *(const v8bf*)(pl + 16));
        }
        // B fragments + WMMA: B 32x16 layout (lane<16: K 0..7 & 16..23; lane>=16: +8)
#pragma unroll
        for (int ni = 0; ni < 4; ++ni) {
            const int colh = waveN * 64 + ni * 16 + l16;
            const int k0   = halfsel * 8;
            const bf16* pb = &Bs_hi[buf][colh * LDB + k0];
            v16bf bhi = cat16(*(const v8bf*)pb, *(const v8bf*)(pb + 16));
            const bf16* pq = &Bs_lo[buf][colh * LDB + k0];
            v16bf blo = cat16(*(const v8bf*)pq, *(const v8bf*)(pq + 16));
#pragma unroll
            for (int mi = 0; mi < 2; ++mi) {
                acc[mi][ni] = __builtin_amdgcn_wmma_f32_16x16x32_bf16(
                    false, ahi[mi], false, bhi, (short)0, acc[mi][ni], false, false);
                acc[mi][ni] = __builtin_amdgcn_wmma_f32_16x16x32_bf16(
                    false, ahi[mi], false, blo, (short)0, acc[mi][ni], false, false);
                acc[mi][ni] = __builtin_amdgcn_wmma_f32_16x16x32_bf16(
                    false, alo[mi], false, bhi, (short)0, acc[mi][ni], false, false);
            }
        }
    };

    // ---- software pipeline: prologue stages tile 0, loop overlaps loads with WMMA ----
    load_tile(0);
    store_tile(0);
    __syncthreads();

    for (int kti = 0; kti < NK; ++kti) {
        const int  cur  = kti & 1;
        const bool more = (kti + 1) < NK;
        if (more) {
            load_tile((kti + 1) * BK);   // global loads in flight during WMMA below
            if (kti + 2 < NK) {          // global_prefetch_b8 two tiles ahead
                __builtin_prefetch(xA + (size_t)aRow0 * DIM_I + (kti + 2) * BK + aCol, 0, 0);
                __builtin_prefetch(Wb + (size_t)((kti + 2) * BK + bRow0) * DIM_H + bCol, 0, 0);
            }
        }
        compute(cur);
        if (more) store_tile(cur ^ 1);
        __syncthreads();
    }

    // ---- epilogue: add bias, nontemporal store. C/D layout: VGPR r -> m = r + 8*(lane>=16), n = lane&15 ----
    float* outp = out + ((size_t)b * DIM_T + t0) * DIM_H + h0;
#pragma unroll
    for (int ni = 0; ni < 4; ++ni) {
        const int ncol = waveN * 64 + ni * 16 + l16;
        const float bv = bias[(size_t)cat * DIM_H + h0 + ncol];
#pragma unroll
        for (int mi = 0; mi < 2; ++mi) {
            const int mbase = waveM * 32 + mi * 16 + halfsel * 8;
#pragma unroll
            for (int r = 0; r < 8; ++r) {
                __builtin_nontemporal_store(acc[mi][ni][r] + bv,
                                            &outp[(size_t)(mbase + r) * DIM_H + ncol]);
            }
        }
    }
}

extern "C" void kernel_launch(void* const* d_in, const int* in_sizes, int n_in,
                              void* d_out, int out_size, void* d_ws, size_t ws_size,
                              hipStream_t stream) {
    const float* x       = (const float*)d_in[0];
    const int*   cat_ids = (const int*)d_in[1];
    const float* W       = (const float*)d_in[2];
    const float* bias    = (const float*)d_in[3];
    float*       out     = (float*)d_out;

    dim3 grid(DIM_H / BN, DIM_T / BM, DIM_B);  // (32, 2, 64)
    dim3 block(256);
    cat_linear_wmma<<<grid, block, 0, stream>>>(x, cat_ids, W, bias, out);
}
